// LinformerAttention_59914793779401
// MI455X (gfx1250) — matrix operand order repaired and checked by reading, not compiled, not run on hardware
//
#include <hip/hip_runtime.h>

// ---------------------------------------------------------------------------
// Linformer attention on MI455X (gfx1250), bf16 WMMA pipeline with
// async-to-LDS double-buffered GEMM (CDNA5 GLOBAL_LOAD_ASYNC_TO_LDS path).
// B=4, N=4096, D=1024, H=16, R=64, HD=64
// ---------------------------------------------------------------------------

typedef __attribute__((ext_vector_type(16))) __bf16 bf16x16;
typedef __attribute__((ext_vector_type(8)))  __bf16 bf16x8;
typedef __attribute__((ext_vector_type(8)))  float  f32x8;

#define WMMA_BF16(a, b, c)                                                   \
  __builtin_amdgcn_wmma_f32_16x16x32_bf16(false, (a), false, (b), (short)0,  \
                                          (c), false, false)

// Build an A-fragment (16x32 bf16) for one lane from a row-major, K-contiguous
// source. rowp must already point at this lane's row. Per ISA 7.12.2:
// lanes 0-15: K = {k0..k0+7, k0+16..k0+23}; lanes 16-31: offset by 8.
__device__ __forceinline__ bf16x16 make_a_frag(const __bf16* rowp, int k0, int hi) {
  const __bf16* p = rowp + k0 + hi * 8;
  bf16x8 lo = *(const bf16x8*)p;
  bf16x8 hh = *(const bf16x8*)(p + 16);
  bf16x16 f;
#pragma unroll
  for (int i = 0; i < 8; ++i) { f[i] = lo[i]; f[i + 8] = hh[i]; }
  return f;
}

// CDNA5 async copy: 16 bytes per lane, global -> LDS, tracked by ASYNCcnt.
// The LDS destination is passed as a pointer so the buffer escapes into the
// asm (keeps the consumer ds_loads alive); for an LDS-aperture flat address,
// addr[31:0] is exactly the LDS byte offset the instruction consumes.
__device__ __forceinline__ void async_copy_b128(void* lds_ptr,
                                                const void* gptr) {
  unsigned lds_off = (unsigned)(size_t)lds_ptr;
  asm volatile("global_load_async_to_lds_b128 %0, %1, off"
               :: "v"(lds_off), "v"(gptr)
               : "memory");
}

// ---------------------------------------------------------------------------
// fp32 -> bf16 elementwise convert
// ---------------------------------------------------------------------------
__global__ void cvt_f32_bf16_k(const float* __restrict__ in,
                               __bf16* __restrict__ out, int n) {
  int i = blockIdx.x * blockDim.x + threadIdx.x;
  if (i < n) out[i] = (__bf16)in[i];
}

// fp32 (rows x cols) -> bf16 transposed (cols x rows); weights are small.
__global__ void transpose_cvt_k(const float* __restrict__ in,
                                __bf16* __restrict__ out, int rows, int cols) {
  int i = blockIdx.x * blockDim.x + threadIdx.x;
  if (i < rows * cols) {
    int r = i / cols, c = i % cols;
    out[(size_t)c * rows + r] = (__bf16)in[i];
  }
}

// ---------------------------------------------------------------------------
// C = A(MxK bf16, row-major) x Bt(KxN bf16, row-major) [+ bias]
// Block: 256 threads = 8 waves. Block tile 128(M) x 128(N), wave tile 64x32.
// Tiles staged into LDS with GLOBAL_LOAD_ASYNC_TO_LDS_B128, double buffered,
// retired with s_wait_asynccnt so the WMMA stream overlaps the next fetch.
// LDS per buffer: A 128x32 (8KB) + B 32x128 (8KB); 32KB total.
// ---------------------------------------------------------------------------
__global__ __launch_bounds__(256) void gemm_wmma_bf16(
    const __bf16* __restrict__ A, const __bf16* __restrict__ Bt,
    __bf16* __restrict__ Cb, float* __restrict__ Cf,
    const float* __restrict__ bias, int M, int N, int K, int out_f32) {
  __shared__ char lds_raw[2 * 16384];  // [buf: 0/16384] A 8KB | B 8KB

  const int t = threadIdx.x;
  const int lane = t & 31;
  const int wave = t >> 5;
  const int l = lane & 15;
  const int hi = lane >> 4;
  const int wm = wave & 1;    // 0..1 -> M offset 0/64
  const int wn = wave >> 1;   // 0..3 -> N offset 0/32/64/96
  const int bm = blockIdx.x * 128;
  const int bn = blockIdx.y * 128;

  // Per-thread staging assignments (2 chunks of A, 2 chunks of B per stage).
  // A tile: 128 rows x 32 cols = 512 x 16B chunks; 4 chunks per row.
  const int ar0 = t >> 2,           ac0 = (t & 3) * 8;
  const int ar1 = (t + 256) >> 2,   ac1 = ((t + 256) & 3) * 8;
  // B tile: 32 rows x 128 cols = 512 x 16B chunks; 16 chunks per row.
  const int br0 = t >> 4,           bc0 = (t & 15) * 8;
  const int br1 = (t + 256) >> 4,   bc1 = ((t + 256) & 15) * 8;

  // Loop-invariant global source pointers (advance by k each stage).
  const __bf16* ga0 = A + (size_t)(bm + ar0) * K + ac0;
  const __bf16* ga1 = A + (size_t)(bm + ar1) * K + ac1;
  const __bf16* gb0 = Bt + (size_t)br0 * N + bn + bc0;
  const __bf16* gb1 = Bt + (size_t)br1 * N + bn + bc1;
  // Loop-invariant LDS destinations (buf selected by +16KB).
  char* la0 = lds_raw + ar0 * 64 + ac0 * 2;
  char* la1 = lds_raw + ar1 * 64 + ac1 * 2;
  char* lb0 = lds_raw + 8192 + br0 * 256 + bc0 * 2;
  char* lb1 = lds_raw + 8192 + br1 * 256 + bc1 * 2;

#define STAGE(buf, kk)                                                       \
  do {                                                                       \
    const unsigned bo = (unsigned)(buf) << 14;                               \
    async_copy_b128(la0 + bo, ga0 + (kk));                                   \
    async_copy_b128(la1 + bo, ga1 + (kk));                                   \
    async_copy_b128(lb0 + bo, gb0 + (size_t)(kk) * N);                       \
    async_copy_b128(lb1 + bo, gb1 + (size_t)(kk) * N);                       \
  } while (0)

  f32x8 acc[4][2] = {};
  const int niter = K / 32;
  STAGE(0, 0);
  for (int i = 0; i < niter; ++i) {
    const int buf = i & 1;
    if (i + 1 < niter) {
      STAGE(buf ^ 1, (i + 1) * 32);
      asm volatile("s_wait_asynccnt 4" ::: "memory");  // buffer `buf` landed
    } else {
      asm volatile("s_wait_asynccnt 0" ::: "memory");
    }
    __syncthreads();

    const __bf16* lA = (const __bf16*)&lds_raw[buf << 14];
    const __bf16* lB = (const __bf16*)&lds_raw[(buf << 14) + 8192];

    bf16x16 afr[4];
#pragma unroll
    for (int mt = 0; mt < 4; ++mt)
      afr[mt] = make_a_frag(lA + (size_t)(wm * 64 + mt * 16 + l) * 32, 0, hi);
    const __bf16* bp = lB + (size_t)(hi * 8) * 128 + wn * 32 + l;  // lane = N column
    bf16x16 b0, b1;
#pragma unroll
    for (int s = 0; s < 8; ++s) {
      b0[s] = bp[s * 128]; b0[s + 8] = bp[(s + 16) * 128];
      b1[s] = bp[s * 128 + 16]; b1[s + 8] = bp[(s + 16) * 128 + 16];
    }
#pragma unroll
    for (int mt = 0; mt < 4; ++mt) {
      acc[mt][0] = WMMA_BF16(afr[mt], b0, acc[mt][0]);
      acc[mt][1] = WMMA_BF16(afr[mt], b1, acc[mt][1]);
    }
    __syncthreads();
  }
#undef STAGE

#pragma unroll
  for (int mt = 0; mt < 4; ++mt)
#pragma unroll
    for (int nt = 0; nt < 2; ++nt)
#pragma unroll
      for (int i = 0; i < 8; ++i) {
        int mm = bm + wm * 64 + mt * 16 + i + hi * 8;
        int nn = bn + wn * 32 + nt * 16 + l;
        float v = acc[mt][nt][i];
        if (out_f32)
          Cf[(size_t)mm * N + nn] = v + bias[nn];
        else
          Cb[(size_t)mm * N + nn] = (__bf16)v;
      }
}

// ---------------------------------------------------------------------------
// Low-rank sequence reduction, one block per (b,h):
//   K_lowT[d][r] = sum_n E[n][r] * K[b,n,h*64+d]   (stored HD x R)
//   V_low [r][d] = sum_n F[n][r] * V[b,n,h*64+d]   (stored R x HD)
// Waves 0-3 do K_low (one 16-row R-tile each), waves 4-7 do V_low.
// E/F tiles staged transposed through LDS so A-fragments read contiguously.
// ---------------------------------------------------------------------------
__global__ __launch_bounds__(256) void lowrank_wmma(
    const __bf16* __restrict__ Eb, const __bf16* __restrict__ Fb,
    const __bf16* __restrict__ Kb, const __bf16* __restrict__ Vb,
    __bf16* __restrict__ KlowT, __bf16* __restrict__ Vlow) {
  __shared__ __bf16 et[64][32];  // [r][n-chunk]
  __shared__ __bf16 ft[64][32];
  const int bh = blockIdx.x;
  const int b = bh >> 4;
  const int h = bh & 15;
  const int t = threadIdx.x;
  const int lane = t & 31;
  const int wave = t >> 5;
  const int l = lane & 15;
  const int hi = lane >> 4;
  const int r0 = (wave & 3) * 16;
  const bool doV = wave >= 4;
  const __bf16* bigp =
      (doV ? Vb : Kb) + (size_t)b * 4096 * 1024 + (size_t)h * 64;

  const int lrow = t >> 3;        // 0..31  (n within tile)
  const int lcol = (t & 7) * 8;   // 0..56  (r)

  f32x8 a0 = {}, a1 = {}, a2 = {}, a3 = {};
  for (int n = 0; n < 4096; n += 32) {
    // cooperative transposed staging of E and F tiles (32n x 64r)
    bf16x8 ve = *(const bf16x8*)(Eb + (size_t)(n + lrow) * 64 + lcol);
    bf16x8 vf = *(const bf16x8*)(Fb + (size_t)(n + lrow) * 64 + lcol);
#pragma unroll
    for (int i = 0; i < 8; ++i) {
      et[lcol + i][lrow] = ve[i];
      ft[lcol + i][lrow] = vf[i];
    }
    __syncthreads();

    const __bf16* arow = (doV ? &ft[0][0] : &et[0][0]) + (r0 + l) * 32;
    bf16x16 af = make_a_frag(arow, 0, hi);
    const __bf16* bp = bigp + (size_t)(n + hi * 8) * 1024 + l;  // lane = d column
    bf16x16 bq[4];
#pragma unroll
    for (int f = 0; f < 4; ++f)
#pragma unroll
      for (int s = 0; s < 8; ++s) {
        bq[f][s] = bp[(size_t)s * 1024 + f * 16];
        bq[f][s + 8] = bp[(size_t)(s + 16) * 1024 + f * 16];
      }
    a0 = WMMA_BF16(af, bq[0], a0);
    a1 = WMMA_BF16(af, bq[1], a1);
    a2 = WMMA_BF16(af, bq[2], a2);
    a3 = WMMA_BF16(af, bq[3], a3);
    __syncthreads();
  }
  f32x8 aa[4] = {a0, a1, a2, a3};
  if (!doV) {
    __bf16* dst = KlowT + (size_t)bh * 4096;  // HD x R
#pragma unroll
    for (int f = 0; f < 4; ++f)
#pragma unroll
      for (int i = 0; i < 8; ++i) {
        int d = f * 16 + l, r = r0 + i + hi * 8;
        dst[(size_t)d * 64 + r] = (__bf16)aa[f][i];
      }
  } else {
    __bf16* dst = Vlow + (size_t)bh * 4096;  // R x HD
#pragma unroll
    for (int f = 0; f < 4; ++f)
#pragma unroll
      for (int i = 0; i < 8; ++i) {
        int d = f * 16 + l, r = r0 + i + hi * 8;
        dst[(size_t)r * 64 + d] = (__bf16)aa[f][i];
      }
  }
}

// ---------------------------------------------------------------------------
// Fused scores -> softmax -> context. One wave handles 16 query rows.
// scores(16x64) = Q(16x64) x K_lowT(64x64);  softmax over R=64;
// out(16x64)    = attn(16x64) x V_low(64x64); written bf16 to Ob.
// ---------------------------------------------------------------------------
__global__ __launch_bounds__(256) void attn_wmma(
    const __bf16* __restrict__ Qb, const __bf16* __restrict__ KlowT,
    const __bf16* __restrict__ Vlow, __bf16* __restrict__ Ob) {
  __shared__ __bf16 smem[8][16][64];  // per-wave attention tile
  const int bh = blockIdx.x;
  const int b = bh >> 4;
  const int h = bh & 15;
  const int lane = threadIdx.x & 31;
  const int wave = threadIdx.x >> 5;
  const int l = lane & 15;
  const int hi = lane >> 4;
  const int row0 = blockIdx.y * 128 + wave * 16;  // query row within sequence

  const __bf16* Kl = KlowT + (size_t)bh * 4096;  // HD x R
  const __bf16* Vl = Vlow + (size_t)bh * 4096;   // R x HD
  const __bf16* arowq =
      Qb + ((size_t)b * 4096 + row0 + l) * 1024 + (size_t)h * 64;

  // scores GEMM: K-dim = HD = 64
  f32x8 s0 = {}, s1 = {}, s2 = {}, s3 = {};
#pragma unroll
  for (int k = 0; k < 64; k += 32) {
    bf16x16 a = make_a_frag(arowq, k, hi);
    const __bf16* bp = Kl + (size_t)(k + hi * 8) * 64 + l;  // lane = r column
    bf16x16 bq[4];
#pragma unroll
    for (int f = 0; f < 4; ++f)
#pragma unroll
      for (int s = 0; s < 8; ++s) {
        bq[f][s] = bp[s * 64 + f * 16];
        bq[f][s + 8] = bp[(s + 16) * 64 + f * 16];
      }
    s0 = WMMA_BF16(a, bq[0], s0);
    s1 = WMMA_BF16(a, bq[1], s1);
    s2 = WMMA_BF16(a, bq[2], s2);
    s3 = WMMA_BF16(a, bq[3], s3);
  }

  // softmax over r (striped across 16-lane halves + 4 fragments)
  const float scale = 0.125f;  // 1/sqrt(HD)
  float pr[4][8];
#pragma unroll
  for (int i = 0; i < 8; ++i) {
    float v0 = s0[i] * scale, v1 = s1[i] * scale;
    float v2 = s2[i] * scale, v3 = s3[i] * scale;
    float m = fmaxf(fmaxf(v0, v1), fmaxf(v2, v3));
#pragma unroll
    for (int d = 1; d < 16; d <<= 1) m = fmaxf(m, __shfl_xor(m, d, 32));
    float e0 = __expf(v0 - m), e1 = __expf(v1 - m);
    float e2 = __expf(v2 - m), e3 = __expf(v3 - m);
    float sm = e0 + e1 + e2 + e3;
#pragma unroll
    for (int d = 1; d < 16; d <<= 1) sm += __shfl_xor(sm, d, 32);
    float inv = 1.0f / sm;
    pr[0][i] = e0 * inv; pr[1][i] = e1 * inv;
    pr[2][i] = e2 * inv; pr[3][i] = e3 * inv;
  }

  // re-stripe C-layout -> A-layout through LDS (wave-private tile)
#pragma unroll
  for (int f = 0; f < 4; ++f)
#pragma unroll
    for (int i = 0; i < 8; ++i)
      smem[wave][i + hi * 8][f * 16 + l] = (__bf16)pr[f][i];
  __syncthreads();

  // context GEMM: K-dim = R = 64
  f32x8 o0 = {}, o1 = {}, o2 = {}, o3 = {};
  const __bf16* arowp = &smem[wave][l][0];
#pragma unroll
  for (int k = 0; k < 64; k += 32) {
    bf16x16 a = make_a_frag(arowp, k, hi);
    const __bf16* bp = Vl + (size_t)(k + hi * 8) * 64 + l;  // lane = d column
    bf16x16 bq[4];
#pragma unroll
    for (int f = 0; f < 4; ++f)
#pragma unroll
      for (int s = 0; s < 8; ++s) {
        bq[f][s] = bp[s * 64 + f * 16];
        bq[f][s + 8] = bp[(s + 16) * 64 + f * 16];
      }
    o0 = WMMA_BF16(a, bq[0], o0);
    o1 = WMMA_BF16(a, bq[1], o1);
    o2 = WMMA_BF16(a, bq[2], o2);
    o3 = WMMA_BF16(a, bq[3], o3);
  }
  f32x8 oo[4] = {o0, o1, o2, o3};
  __bf16* orow = Ob + ((size_t)b * 4096 + row0) * 1024 + (size_t)h * 64;
#pragma unroll
  for (int f = 0; f < 4; ++f)
#pragma unroll
    for (int i = 0; i < 8; ++i)
      orow[(size_t)(i + hi * 8) * 1024 + f * 16 + l] = (__bf16)oo[f][i];
}

// ---------------------------------------------------------------------------
// Host-side orchestration
// ---------------------------------------------------------------------------
extern "C" void kernel_launch(void* const* d_in, const int* in_sizes, int n_in,
                              void* d_out, int out_size, void* d_ws,
                              size_t ws_size, hipStream_t stream) {
  (void)in_sizes; (void)n_in; (void)out_size; (void)ws_size;
  const float* x  = (const float*)d_in[0];
  const float* Wq = (const float*)d_in[1];
  const float* Wk = (const float*)d_in[2];
  const float* Wv = (const float*)d_in[3];
  const float* Wo = (const float*)d_in[4];
  const float* bo = (const float*)d_in[5];
  const float* E  = (const float*)d_in[6];
  const float* Fm = (const float*)d_in[7];
  float* out = (float*)d_out;

  const size_t MD = (size_t)16384 * 1024;  // B*N x D
  const size_t DD = (size_t)1024 * 1024;
  const size_t NR = (size_t)4096 * 64;
  const size_t LOW = (size_t)64 * 64 * 64;  // B*H x R x HD

  char* ws = (char*)d_ws;
  size_t off = 0;
  __bf16* xb  = (__bf16*)(ws + off); off += MD * 2;
  __bf16* WqT = (__bf16*)(ws + off); off += DD * 2;
  __bf16* WkT = (__bf16*)(ws + off); off += DD * 2;
  __bf16* WvT = (__bf16*)(ws + off); off += DD * 2;
  __bf16* WoT = (__bf16*)(ws + off); off += DD * 2;
  __bf16* Eb  = (__bf16*)(ws + off); off += NR * 2;
  __bf16* Fb  = (__bf16*)(ws + off); off += NR * 2;
  __bf16* Qb  = (__bf16*)(ws + off); off += MD * 2;
  __bf16* Kb  = (__bf16*)(ws + off); off += MD * 2;
  __bf16* Vb  = (__bf16*)(ws + off); off += MD * 2;
  __bf16* KlowT = (__bf16*)(ws + off); off += LOW * 2;
  __bf16* VlowB = (__bf16*)(ws + off); off += LOW * 2;
  __bf16* Ob  = (__bf16*)(ws + off); off += MD * 2;

  // 1) convert inputs to bf16 (weights transposed to K-major for B-fragments)
  cvt_f32_bf16_k<<<(int)((MD + 1023) / 1024), 1024, 0, stream>>>(x, xb, (int)MD);
  transpose_cvt_k<<<(int)((DD + 255) / 256), 256, 0, stream>>>(Wq, WqT, 1024, 1024);
  transpose_cvt_k<<<(int)((DD + 255) / 256), 256, 0, stream>>>(Wk, WkT, 1024, 1024);
  transpose_cvt_k<<<(int)((DD + 255) / 256), 256, 0, stream>>>(Wv, WvT, 1024, 1024);
  transpose_cvt_k<<<(int)((DD + 255) / 256), 256, 0, stream>>>(Wo, WoT, 1024, 1024);
  cvt_f32_bf16_k<<<(int)((NR + 255) / 256), 256, 0, stream>>>(E, Eb, (int)NR);
  cvt_f32_bf16_k<<<(int)((NR + 255) / 256), 256, 0, stream>>>(Fm, Fb, (int)NR);

  // 2) Q/K/V projections: (16384 x 1024) x (1024 x 1024)
  dim3 ggrid(16384 / 128, 1024 / 128);
  gemm_wmma_bf16<<<ggrid, 256, 0, stream>>>(xb, WqT, Qb, nullptr, bo,
                                            16384, 1024, 1024, 0);
  gemm_wmma_bf16<<<ggrid, 256, 0, stream>>>(xb, WkT, Kb, nullptr, bo,
                                            16384, 1024, 1024, 0);
  gemm_wmma_bf16<<<ggrid, 256, 0, stream>>>(xb, WvT, Vb, nullptr, bo,
                                            16384, 1024, 1024, 0);

  // 3) low-rank sequence reductions per (b,h)
  lowrank_wmma<<<64, 256, 0, stream>>>(Eb, Fb, Kb, Vb, KlowT, VlowB);

  // 4) fused scores/softmax/context
  attn_wmma<<<dim3(64, 4096 / 128), 256, 0, stream>>>(Qb, KlowT, VlowB, Ob);

  // 5) output projection with bias -> f32 d_out
  gemm_wmma_bf16<<<ggrid, 256, 0, stream>>>(Ob, WoT, nullptr, out, bo,
                                            16384, 1024, 1024, 1);
}
